// memoryModule_67628555042904
// MI455X (gfx1250) — matrix operand order repaired and checked by reading, not compiled, not run on hardware
//
#include <hip/hip_runtime.h>

typedef __attribute__((ext_vector_type(16))) _Float16 v16h;
typedef __attribute__((ext_vector_type(8)))  _Float16 v8h;
typedef __attribute__((ext_vector_type(8)))  float    v8f;
typedef __attribute__((ext_vector_type(4)))  float    v4f;

#define B_   64
#define C_   128
#define HW_  16384
#define L_   50
#define LP_  64
#define TP_  128      // pixels per workgroup (8 waves x 16)
#define XP_  136      // x-tile pitch in halfs (16B-aligned rows, bank-skewed)

// dynamic LDS layout (byte offsets, all 16B multiples)
#define OFF_KN    0        // knorm  [64][128] f16   (16384B)
#define OFF_VT    16384    // valuesT[128][64] f16   (16384B)
#define OFF_XT    32768    // xtile  [128][XP_] f16  (34816B)
#define OFF_INVN  67584    // invnorm[128] f32       (512B)
#define OFF_KINV  68096    // key invnorm[64] f32    (256B)
#define OFF_PSUM  68352    // per-thread sumsq [256] (1024B)
#define OFF_KRAW  69376    // raw keys   [50][128] f32 (25600B, async-DMA'd)
#define OFF_VRAW  94976    // raw values [50][128] f32 (25600B, async-DMA'd)
#define SMEM_BYTES 120576

__global__ void __launch_bounds__(256)
memory_module_kernel(const float* __restrict__ x,
                     const float* __restrict__ keys,
                     const float* __restrict__ values,
                     float* __restrict__ out)
{
    extern __shared__ char smem[];
    _Float16* kn   = (_Float16*)(smem + OFF_KN);
    _Float16* vT   = (_Float16*)(smem + OFF_VT);
    _Float16* xt   = (_Float16*)(smem + OFF_XT);
    float*    invn = (float*)(smem + OFF_INVN);
    float*    kinv = (float*)(smem + OFF_KINV);
    float*    psum = (float*)(smem + OFF_PSUM);
    float*    kraw = (float*)(smem + OFF_KRAW);
    float*    vraw = (float*)(smem + OFF_VRAW);

    const int tid = threadIdx.x;

    // ---- Phase 0: async DMA raw keys/values into LDS (ASYNCcnt path) ----
    // 50*128 f32 = 25600B = 1600 x 16B chunks each; per-lane 16B async copies.
    #pragma unroll
    for (int it = 0; it < 7; ++it) {
        int chunk = it * 256 + tid;
        if (chunk < 1600) {
            unsigned     lk = (unsigned)(size_t)(kraw + chunk * 4);
            const float* gk = keys + chunk * 4;
            asm volatile("global_load_async_to_lds_b128 %0, %1, off"
                         :: "v"(lk), "v"(gk) : "memory");
            unsigned     lv = (unsigned)(size_t)(vraw + chunk * 4);
            const float* gv = values + chunk * 4;
            asm volatile("global_load_async_to_lds_b128 %0, %1, off"
                         :: "v"(lv), "v"(gv) : "memory");
        }
    }
    asm volatile("s_wait_asynccnt 0" ::: "memory");
    __syncthreads();

    // ---- Phase 1: key norms (4 lanes per slot, intra-wave shfl reduce) ----
    {
        int l = tid >> 2, q = tid & 3;
        float s = 0.f;
        if (l < L_) {
            const float* kp = kraw + l * C_ + q * 32;
            #pragma unroll
            for (int i = 0; i < 32; ++i) { float v = kp[i]; s += v * v; }
        }
        s += __shfl_xor(s, 1, 32);
        s += __shfl_xor(s, 2, 32);
        if (q == 0 && l < L_) kinv[l] = 1.f / fmaxf(sqrtf(s), 1e-12f);
    }
    __syncthreads();

    // ---- Phase 2: stage normalized keys + transposed values as f16 ----
    #pragma unroll
    for (int it = 0; it < 32; ++it) {
        int idx = it * 256 + tid;                  // 8192 elements each
        int l  = idx >> 7, c  = idx & 127;
        kn[idx] = (l < L_) ? (_Float16)(kraw[l * C_ + c] * kinv[l]) : (_Float16)0.f;
        int c2 = idx >> 6, l2 = idx & 63;
        vT[idx] = (l2 < L_) ? (_Float16)vraw[l2 * C_ + c2] : (_Float16)0.f;
    }

    // ---- Phase 3: stream x tile, fused sum-of-squares, f16 transpose ----
    const int tile = blockIdx.x;
    const int b    = (tile * TP_) >> 14;           // image index
    const int pim0 = (tile * TP_) & (HW_ - 1);     // pixel base within image
    const float* xb = x + ((size_t)b * C_) * HW_ + pim0;

    {
        const int p  = tid & 127;                  // this thread's pixel
        const int hi = tid >> 7;
        float sacc = 0.f;
        #pragma unroll 4
        for (int it = 0; it < 64; ++it) {
            int c = it * 2 + hi;
            int cpf = c + 16;
            if (cpf < C_) __builtin_prefetch(xb + (size_t)cpf * HW_ + p, 0, 0);
            float v = xb[(size_t)c * HW_ + p];     // coalesced across lanes
            sacc += v * v;
            xt[p * XP_ + c] = (_Float16)v;
        }
        psum[tid] = sacc;
    }
    __syncthreads();
    if (tid < TP_) {
        float s = psum[tid] + psum[tid + 128];
        invn[tid] = 1.f / fmaxf(sqrtf(s), 1e-12f);
    }
    __syncthreads();

    // ---- Phase 4: GEMM1  cos[l, p] = knorm . x  (scale by invn later) ----
    const int lane = tid & 31;
    const int wave = tid >> 5;
    const int col  = lane & 15;
    const int hiL  = lane >> 4;                    // 0 or 1
    const int p    = wave * 16 + col;              // pixel column (tile-local)

    v8f acc[4] = {};
    #pragma unroll
    for (int ks = 0; ks < 4; ++ks) {
        // B fragment: channels {kb..kb+7, kb+16..kb+23} for this pixel
        const _Float16* bp = xt + p * XP_ + ks * 32 + hiL * 8;
        v8h b0 = *(const v8h*)(bp);
        v8h b1 = *(const v8h*)(bp + 16);
        v16h bf;
        #pragma unroll
        for (int i = 0; i < 8; ++i) { bf[i] = b0[i]; bf[i + 8] = b1[i]; }
        #pragma unroll
        for (int mt = 0; mt < 4; ++mt) {
            // A fragment: knorm row l = mt*16+col, K chunks {ka..ka+7, ka+16..ka+23}
            const _Float16* ap = kn + (mt * 16 + col) * C_ + ks * 32 + hiL * 8;
            v8h a0 = *(const v8h*)(ap);
            v8h a1 = *(const v8h*)(ap + 16);
            v16h af;
            #pragma unroll
            for (int i = 0; i < 8; ++i) { af[i] = a0[i]; af[i + 8] = a1[i]; }
            acc[mt] = __builtin_amdgcn_wmma_f32_16x16x32_f16(
                false, af, false, bf, (short)0, acc[mt], false, false);
        }
    }

    // ---- Phase 5: softmax over 50 slots (rows split across lane, lane^16) ----
    const float inv = invn[p];
    float sv[4][8];
    float m = -3.4e38f;
    #pragma unroll
    for (int t = 0; t < 4; ++t)
        #pragma unroll
        for (int v = 0; v < 8; ++v) {
            int l = t * 16 + hiL * 8 + v;
            float cv = acc[t][v] * inv;
            sv[t][v] = cv;
            if (l < L_) m = fmaxf(m, cv);
        }
    m = fmaxf(m, __shfl_xor(m, 16, 32));
    float s = 0.f;
    #pragma unroll
    for (int t = 0; t < 4; ++t)
        #pragma unroll
        for (int v = 0; v < 8; ++v) {
            int l = t * 16 + hiL * 8 + v;
            float e = (l < L_) ? __expf(sv[t][v] - m) : 0.f;
            sv[t][v] = e;
            s += e;
        }
    s += __shfl_xor(s, 16, 32);
    const float r = 1.f / s;

    // GEMM1-D layout == GEMM2-A layout: pack probabilities in-register
    v16h a2[2];
    #pragma unroll
    for (int ks = 0; ks < 2; ++ks)
        #pragma unroll
        for (int j = 0; j < 16; ++j)
            a2[ks][j] = (_Float16)(sv[ks * 2 + (j >> 3)][j & 7] * r);

    // ---- Phase 6: GEMM2  fnorm[p, c] = sim^T . values ; direct b128 stores ----
    float* ob = out + ((size_t)b * C_) * HW_ + pim0 + wave * 16 + hiL * 8;
    #pragma unroll
    for (int nt = 0; nt < 8; ++nt) {
        v8f d = {};
        #pragma unroll
        for (int ks = 0; ks < 2; ++ks) {
            const _Float16* bp2 = vT + (nt * 16 + col) * LP_ + ks * 32 + hiL * 8;
            v8h b0 = *(const v8h*)(bp2);
            v8h b1 = *(const v8h*)(bp2 + 16);
            v16h bf;
            #pragma unroll
            for (int i = 0; i < 8; ++i) { bf[i] = b0[i]; bf[i + 8] = b1[i]; }
            d = __builtin_amdgcn_wmma_f32_16x16x32_f16(
                false, a2[ks], false, bf, (short)0, d, false, false);
        }
        int c = nt * 16 + col;
        float* o = ob + (size_t)c * HW_;
        v4f lo = {d[0], d[1], d[2], d[3]};
        v4f hi4 = {d[4], d[5], d[6], d[7]};
        *(v4f*)(o)     = lo;                       // 8 consecutive pixels per lane
        *(v4f*)(o + 4) = hi4;                      // -> 2x global_store_b128
    }
}

extern "C" void kernel_launch(void* const* d_in, const int* in_sizes, int n_in,
                              void* d_out, int out_size, void* d_ws, size_t ws_size,
                              hipStream_t stream) {
    (void)in_sizes; (void)n_in; (void)out_size; (void)d_ws; (void)ws_size;
    const float* x      = (const float*)d_in[0];
    const float* keys   = (const float*)d_in[1];
    const float* values = (const float*)d_in[2];
    float* out = (float*)d_out;

    (void)hipFuncSetAttribute(reinterpret_cast<const void*>(memory_module_kernel),
                              hipFuncAttributeMaxDynamicSharedMemorySize, SMEM_BYTES);

    dim3 grid((B_ * HW_) / TP_);   // 8192 workgroups
    memory_module_kernel<<<grid, 256, SMEM_BYTES, stream>>>(x, keys, values, out);
}
